// LSTMActor_14353780703463
// MI455X (gfx1250) — matrix operand order repaired and checked by reading, not compiled, not run on hardware
//
#include <hip/hip_runtime.h>
#include <hip/hip_bf16.h>

// ---------------------------------------------------------------------------
// LSTM actor for MI455X (gfx1250, wave32).
// Compute-bound workload (AI ~500 FLOP/B): use v_wmma_f32_16x16x32_f16 for all
// GEMMs (batch tile of 16 = WMMA M), native v_tanh_f32 (CDNA5 transcendental)
// for sigmoid/tanh, LDS f16 staging between layers, and a one-shot prep
// kernel that pre-swizzles fc1/fc2 weights into per-lane B-fragment layout in
// d_ws for fully coalesced 32B-per-lane fragment loads.
// ---------------------------------------------------------------------------

typedef __attribute__((ext_vector_type(16))) _Float16 v16h;
typedef __attribute__((ext_vector_type(8)))  _Float16 v8h;
typedef __attribute__((ext_vector_type(8)))  float    v8f;

#define B_TOT   262144
#define T_STEPS 24
#define D_IN    3
#define U_IN    2
#define H1      8
#define H2      16
#define FC1_OUT 128
#define ACT_OUT 24
#define OBS_W   (D_IN * T_STEPS + U_IN)   // 74
#define FEAT_K  (T_STEPS * H2)            // 384
#define FC1_IN  (FEAT_K + U_IN)           // 386
#define FC1_KC  (FEAT_K / 32)             // 12 k-chunks of 32
#define FC1_NT  (FC1_OUT / 16)            // 8 n-tiles
#define FC2_KC  (FC1_OUT / 32)            // 4
#define FC2_NT  2                         // 24 -> 2 n-tiles (pad to 32)

#define FC1_FRAG_HALVES (FC1_KC * FC1_NT * 32 * 16)  // 49152
#define FC2_FRAG_HALVES (FC2_KC * FC2_NT * 32 * 16)  // 4096

// ---- fast transcendentals -------------------------------------------------
__device__ __forceinline__ float frcp(float x)  { return __builtin_amdgcn_rcpf(x); }

#if __has_builtin(__builtin_amdgcn_tanhf)
// CDNA5 has native V_TANH_F32 (ISA §8.4 lists TANH among TRANS32 ops).
__device__ __forceinline__ float ftanh_(float x) { return __builtin_amdgcn_tanhf(x); }
// sigmoid(x) = 0.5 + 0.5*tanh(x/2): 1 trans + 2 VALU instead of exp+add+rcp.
__device__ __forceinline__ float fsig(float x) {
  return __builtin_fmaf(0.5f, __builtin_amdgcn_tanhf(0.5f * x), 0.5f);
}
#else
__device__ __forceinline__ float fexp(float x)  { return __expf(x); }
__device__ __forceinline__ float fsig(float x)  { return frcp(1.f + fexp(-x)); }
__device__ __forceinline__ float ftanh_(float x){ return 1.f - 2.f * frcp(1.f + fexp(2.f * x)); }
#endif

__device__ __forceinline__ v8f wmma16(v16h a, v16h b, v8f c) {
  // (neg_a, A, neg_b, B, c_mod, C, reuse_a, reuse_b)
  return __builtin_amdgcn_wmma_f32_16x16x32_f16(false, a, false, b, (short)0, c,
                                                false, false);
}

// ---------------------------------------------------------------------------
// Prep kernel: swizzle fc1_w/fc2_w (f32, row-major [n][k]) into per-lane f16
// B-fragment layout:
//   fragment (kc, nt), lane l, element i  ->  W[k = kc*32 + (i<8?0:16) + (l<16?0:8) + (i&7)]
//                                              [n = nt*16 + (l&15)]
// so the main kernel loads one contiguous v16h (32B) per lane per fragment.
// ---------------------------------------------------------------------------
__global__ void prep_weights(const float* __restrict__ fc1_w,
                             const float* __restrict__ fc2_w,
                             _Float16* __restrict__ frag) {
  int tid = blockIdx.x * blockDim.x + threadIdx.x;
  if (tid < FC1_FRAG_HALVES) {
    int i    = tid & 15;
    int lane = (tid >> 4) & 31;
    int nt   = (tid >> 9) & 7;
    int kc   = tid >> 12;
    int k = kc * 32 + ((i < 8) ? 0 : 16) + ((lane < 16) ? 0 : 8) + (i & 7);     // < 384 < 386
    int n = nt * 16 + (lane & 15);
    frag[tid] = (_Float16)fc1_w[n * FC1_IN + k];
  } else if (tid < FC1_FRAG_HALVES + FC2_FRAG_HALVES) {
    int t2   = tid - FC1_FRAG_HALVES;
    int i    = t2 & 15;
    int lane = (t2 >> 4) & 31;
    int nt   = (t2 >> 9) & 1;
    int kc   = t2 >> 10;
    int k = kc * 32 + ((i < 8) ? 0 : 16) + ((lane < 16) ? 0 : 8) + (i & 7);     // < 128
    int n = nt * 16 + (lane & 15);
    frag[tid] = (n < ACT_OUT) ? (_Float16)fc2_w[n * FC1_OUT + k] : (_Float16)0.f;
  }
}

// ---------------------------------------------------------------------------
// Main kernel: 64 threads = 2 waves, each wave owns a tile of 16 batch rows.
// ---------------------------------------------------------------------------
__launch_bounds__(64)
__global__ void lstm_actor_kernel(
    const float* __restrict__ obs,
    const float* __restrict__ w_ih1, const float* __restrict__ w_hh1,
    const float* __restrict__ b_ih1, const float* __restrict__ b_hh1,
    const float* __restrict__ w_ih2, const float* __restrict__ w_hh2,
    const float* __restrict__ b_ih2, const float* __restrict__ b_hh2,
    const float* __restrict__ fc1_w, const float* __restrict__ fc1_b,
    const float* __restrict__ fc2_b,
    const _Float16* __restrict__ wfrag,
    float* __restrict__ out) {

  const int w  = threadIdx.x >> 5;   // wave in block
  const int l  = threadIdx.x & 31;   // lane
  const int j  = l & 15;             // N / column index inside 16-wide tile
  const int hi = (l >> 4) & 1;       // lane half (selects K sub-ranges / M+8)
  const int b0 = (blockIdx.x * 2 + w) * 16;

  // LDS: 38.25 KB / block
  __shared__ __align__(16) _Float16 feats_s[2][16][FEAT_K];   // h2 history (fc1 A)
  __shared__ __align__(16) _Float16 act_s  [2][16][FC1_OUT];  // relu(fc1) (fc2 A)
  __shared__ __align__(16) _Float16 x_s    [2][16][72];       // obs x-part, f16
  __shared__              float     x2_s   [2][16][U_IN];
  __shared__ __align__(16) _Float16 h1_s   [2][16][H1];
  __shared__ __align__(16) _Float16 h2_s   [2][16][H2];

  // ---- stage this wave's 16 obs rows into LDS (coalesced) ----
  for (int idx = l; idx < 16 * OBS_W; idx += 32) {
    int r = idx / OBS_W, c = idx % OBS_W;
    float v = obs[(size_t)(b0 + r) * OBS_W + c];
    if (c < 72) x_s[w][r][c] = (_Float16)v;
    else        x2_s[w][r][c - 72] = v;
  }
  for (int idx = l; idx < 16 * H1; idx += 32) h1_s[w][idx >> 3][idx & 7]  = (_Float16)0.f;
  for (int idx = l; idx < 16 * H2; idx += 32) h2_s[w][idx >> 4][idx & 15] = (_Float16)0.f;

  // ---- build resident LSTM weight B-fragments (once per wave) ----
  // Layer1: K = [x(3), h1(8)] padded to 32; N = 32 gates -> 2 fragments.
  v16h bw1[2];
  #pragma unroll
  for (int nt = 0; nt < 2; ++nt) {
    int n = nt * 16 + j;
    #pragma unroll
    for (int i = 0; i < 16; ++i) {
      int k = ((i < 8) ? 0 : 16) + (hi ? 8 : 0) + (i & 7);
      float v = 0.f;
      if (k < 3)       v = w_ih1[n * D_IN + k];
      else if (k < 11) v = w_hh1[n * H1 + (k - 3)];
      bw1[nt][i] = (_Float16)v;
    }
  }
  // Layer2: K = [h1(8), h2(16)] padded to 32; N = 64 gates -> 4 fragments.
  v16h bw2[4];
  #pragma unroll
  for (int q = 0; q < 4; ++q) {
    int n = q * 16 + j;
    #pragma unroll
    for (int i = 0; i < 16; ++i) {
      int k = ((i < 8) ? 0 : 16) + (hi ? 8 : 0) + (i & 7);
      float v = 0.f;
      if (k < 8)       v = w_ih2[n * H1 + k];
      else if (k < 24) v = w_hh2[n * H2 + (k - 8)];
      bw2[q][i] = (_Float16)v;
    }
  }
  float b1s0 = b_ih1[j]      + b_hh1[j];
  float b1s1 = b_ih1[16 + j] + b_hh1[16 + j];
  float b2s[4];
  #pragma unroll
  for (int q = 0; q < 4; ++q) b2s[q] = b_ih2[q * 16 + j] + b_hh2[q * 16 + j];

  float c1r[8], c2r[8];
  #pragma unroll
  for (int r = 0; r < 8; ++r) { c1r[r] = 0.f; c2r[r] = 0.f; }
  __syncthreads();

  // =========================== LSTM time loop ==============================
  for (int t = 0; t < T_STEPS; ++t) {
    // ---- layer 1: A = [x_t | h1] (16 x 32 f16) ----
    v16h a1;
    #pragma unroll
    for (int i = 0; i < 16; ++i) a1[i] = (_Float16)0.f;
    if (!hi) {                     // lanes 0-15: K 0..7 (x0..x2,h1_0..4), 16..23 pad
      a1[0] = x_s[w][j][t];
      a1[1] = x_s[w][j][24 + t];
      a1[2] = x_s[w][j][48 + t];
      #pragma unroll
      for (int q = 0; q < 5; ++q) a1[3 + q] = h1_s[w][j][q];
    } else {                       // lanes 16-31: K 8..10 = h1_5..7, rest pad
      #pragma unroll
      for (int q = 0; q < 3; ++q) a1[q] = h1_s[w][j][5 + q];
    }
    v8f cb0, cb1;
    #pragma unroll
    for (int r = 0; r < 8; ++r) { cb0[r] = b1s0; cb1[r] = b1s1; }
    v8f g0 = wmma16(a1, bw1[0], cb0);   // gates i(N0..7)  f(N8..15)
    v8f g1 = wmma16(a1, bw1[1], cb1);   // gates g(N16..23) o(N24..31)

    #pragma unroll
    for (int r = 0; r < 8; ++r) {
      float iv = g0[r];
      float fv = __shfl_xor(g0[r], 8, 32);   // bring f[m,j] to lane j
      float gv = g1[r];
      float ov = __shfl_xor(g1[r], 8, 32);   // bring o[m,j] to lane j
      float cn = fsig(fv) * c1r[r] + fsig(iv) * ftanh_(gv);
      float hn = fsig(ov) * ftanh_(cn);
      c1r[r] = cn;
      if (j < 8) h1_s[w][r + hi * 8][j] = (_Float16)hn;
    }
    __syncthreads();

    // ---- layer 2: A = [h1 | h2] (16 x 32 f16) ----
    v16h a2;
    if (!hi) {                     // K 0..7 = h1, K 16..23 = h2[8..15]
      #pragma unroll
      for (int i = 0; i < 8; ++i) a2[i]     = h1_s[w][j][i];
      #pragma unroll
      for (int i = 0; i < 8; ++i) a2[8 + i] = h2_s[w][j][8 + i];
    } else {                       // K 8..15 = h2[0..7], K 24..31 pad
      #pragma unroll
      for (int i = 0; i < 8; ++i) a2[i]     = h2_s[w][j][i];
      #pragma unroll
      for (int i = 0; i < 8; ++i) a2[8 + i] = (_Float16)0.f;
    }
    v8f gg[4];
    #pragma unroll
    for (int q = 0; q < 4; ++q) {
      v8f cb;
      #pragma unroll
      for (int r = 0; r < 8; ++r) cb[r] = b2s[q];
      gg[q] = wmma16(a2, bw2[q], cb);
    }
    #pragma unroll
    for (int r = 0; r < 8; ++r) {
      float iv = gg[0][r], fv = gg[1][r], gv = gg[2][r], ov = gg[3][r];
      float cn = fsig(fv) * c2r[r] + fsig(iv) * ftanh_(gv);
      float hn = fsig(ov) * ftanh_(cn);
      c2r[r] = cn;
      _Float16 hh = (_Float16)hn;
      int m = r + hi * 8;
      h2_s[w][m][j] = hh;
      feats_s[w][m][t * 16 + j] = hh;   // fc1 input, already A-friendly
    }
    __syncthreads();
  }

  // ============================ fc1 (K=386) ================================
  // Bias + x2 tail columns folded into accumulator init.
  float x2a[8], x2b[8];
  #pragma unroll
  for (int r = 0; r < 8; ++r) {
    int m = r + hi * 8;
    x2a[r] = x2_s[w][m][0];
    x2b[r] = x2_s[w][m][1];
  }
  v8f acc[FC1_NT];
  #pragma unroll
  for (int nt = 0; nt < FC1_NT; ++nt) {
    int n = nt * 16 + j;
    float bb = fc1_b[n];
    float wa = fc1_w[n * FC1_IN + 384];
    float wb = fc1_w[n * FC1_IN + 385];
    #pragma unroll
    for (int r = 0; r < 8; ++r) acc[nt][r] = bb + x2a[r] * wa + x2b[r] * wb;
  }
  for (int kc = 0; kc < FC1_KC; ++kc) {
    int off = kc * 32 + (hi ? 8 : 0);
    v8h lo = *(const v8h*)&feats_s[w][j][off];        // ds_load_b128
    v8h hh = *(const v8h*)&feats_s[w][j][off + 16];   // ds_load_b128
    v16h a;
    #pragma unroll
    for (int i = 0; i < 8; ++i) { a[i] = lo[i]; a[8 + i] = hh[i]; }
    #pragma unroll
    for (int nt = 0; nt < FC1_NT; ++nt) {
      v16h bfr = *(const v16h*)&wfrag[((kc * FC1_NT + nt) * 32 + l) * 16];
      acc[nt] = wmma16(a, bfr, acc[nt]);
    }
  }
  // relu -> act_s (A layout for fc2)
  #pragma unroll
  for (int nt = 0; nt < FC1_NT; ++nt) {
    #pragma unroll
    for (int r = 0; r < 8; ++r) {
      float v = acc[nt][r];
      act_s[w][r + hi * 8][nt * 16 + j] = (_Float16)(v > 0.f ? v : 0.f);
    }
  }
  __syncthreads();

  // ============================ fc2 (K=128, N=24) ==========================
  v8f oacc[FC2_NT];
  #pragma unroll
  for (int nt = 0; nt < FC2_NT; ++nt) {
    int n = nt * 16 + j;
    float bb = (n < ACT_OUT) ? fc2_b[n] : 0.f;
    #pragma unroll
    for (int r = 0; r < 8; ++r) oacc[nt][r] = bb;
  }
  #pragma unroll
  for (int kc = 0; kc < FC2_KC; ++kc) {
    int off = kc * 32 + (hi ? 8 : 0);
    v8h lo = *(const v8h*)&act_s[w][j][off];
    v8h hh = *(const v8h*)&act_s[w][j][off + 16];
    v16h a;
    #pragma unroll
    for (int i = 0; i < 8; ++i) { a[i] = lo[i]; a[8 + i] = hh[i]; }
    #pragma unroll
    for (int nt = 0; nt < FC2_NT; ++nt) {
      v16h bfr = *(const v16h*)&wfrag[FC1_FRAG_HALVES +
                                      ((kc * FC2_NT + nt) * 32 + l) * 16];
      oacc[nt] = wmma16(a, bfr, oacc[nt]);
    }
  }
  // soft_sign + store
  #pragma unroll
  for (int nt = 0; nt < FC2_NT; ++nt) {
    int n = nt * 16 + j;
    if (n < ACT_OUT) {
      #pragma unroll
      for (int r = 0; r < 8; ++r) {
        float y = oacc[nt][r];
        y = y * frcp(1.f + fabsf(y));
        out[(size_t)(b0 + r + hi * 8) * ACT_OUT + n] = y;
      }
    }
  }
}

// ---------------------------------------------------------------------------
extern "C" void kernel_launch(void* const* d_in, const int* in_sizes, int n_in,
                              void* d_out, int out_size, void* d_ws, size_t ws_size,
                              hipStream_t stream) {
  const float* obs   = (const float*)d_in[0];
  const float* w_ih1 = (const float*)d_in[1];
  const float* w_hh1 = (const float*)d_in[2];
  const float* b_ih1 = (const float*)d_in[3];
  const float* b_hh1 = (const float*)d_in[4];
  const float* w_ih2 = (const float*)d_in[5];
  const float* w_hh2 = (const float*)d_in[6];
  const float* b_ih2 = (const float*)d_in[7];
  const float* b_hh2 = (const float*)d_in[8];
  const float* fc1_w = (const float*)d_in[9];
  const float* fc1_b = (const float*)d_in[10];
  const float* fc2_w = (const float*)d_in[11];
  const float* fc2_b = (const float*)d_in[12];
  _Float16* frag = (_Float16*)d_ws;   // 104 KB of scratch used

  int prep_elems = FC1_FRAG_HALVES + FC2_FRAG_HALVES;
  prep_weights<<<(prep_elems + 255) / 256, 256, 0, stream>>>(fc1_w, fc2_w, frag);

  // one wave per 16 batch rows, 2 waves per block
  lstm_actor_kernel<<<B_TOT / 32, 64, 0, stream>>>(
      obs, w_ih1, w_hh1, b_ih1, b_hh1, w_ih2, w_hh2, b_ih2, b_hh2,
      fc1_w, fc1_b, fc2_b, frag, (float*)d_out);
}
